// KAGCN_6640019439796
// MI455X (gfx1250) — matrix operand order repaired and checked by reading, not compiled, not run on hardware
//
#include <hip/hip_runtime.h>
#include <hip/hip_bf16.h>

#define NFEAT 128
#define KEXP  1024          // 128 feats * (1 silu + 7 spline bases)
#define NBASES 7
#define NGRAPHS 64
#define NCLASSES 10
#define KC 128              // K-chunk staged in LDS
#define BSTRIDE 136         // padded LDS stride (bf16 elems) per column: 272 B
#define BUFELEMS (64 * BSTRIDE)

#if defined(__has_builtin)
#if __has_builtin(__builtin_amdgcn_global_load_async_to_lds_b128) && \
    __has_builtin(__builtin_amdgcn_s_wait_asynccnt)
#define HAVE_ASYNC_LDS 1
#endif
#endif
#ifndef HAVE_ASYNC_LDS
#define HAVE_ASYNC_LDS 0
#endif

typedef __attribute__((ext_vector_type(8)))  __bf16 v8bf;
typedef __attribute__((ext_vector_type(16))) __bf16 v16bf;
typedef __attribute__((ext_vector_type(8)))  float  v8f;
typedef int v4i __attribute__((vector_size(16)));
typedef __attribute__((address_space(1))) v4i* gptr_v4i;   // global src for async LDS load
typedef __attribute__((address_space(3))) v4i* lptr_v4i;   // LDS dst for async LDS load

__device__ __forceinline__ float silu_f(float x) { return x / (1.0f + expf(-x)); }

__device__ __forceinline__ unsigned short f2bf(float f) {
  union { float f; unsigned u; } v; v.f = f;
  unsigned u = v.u;
  return (unsigned short)((u + 0x7FFFu + ((u >> 16) & 1u)) >> 16);
}

// grid = arange(-3, 8) * 0.5 - 1  -> grid[t] = -2.5 + 0.5*t, t = 0..10
__device__ __forceinline__ void spline_bases(float x, float* bs) {
  const float g0 = -2.5f, h = 0.5f;
  float b[10];
#pragma unroll
  for (int t = 0; t < 10; ++t) {
    float gl = g0 + h * t;
    b[t] = (x >= gl && x < gl + h) ? 1.0f : 0.0f;
  }
#pragma unroll
  for (int k = 1; k <= 3; ++k) {
    float inv = 1.0f / (h * k);
#pragma unroll
    for (int t = 0; t < 10 - k; ++t) {
      float gt = g0 + h * t;
      float left  = (x - gt) * inv;
      float right = ((gt + h * (k + 1)) - x) * inv;
      b[t] = left * b[t] + right * b[t + 1];
    }
  }
#pragma unroll
  for (int t = 0; t < NBASES; ++t) bs[t] = b[t];
}

// Expand: per (node, feat) -> 8 bf16 values [silu(x), bases0..6].
// If bias != null, input is pre-activation: x = silu(raw + bias).
__global__ void expand_kernel(const float* __restrict__ raw,
                              const float* __restrict__ bias,
                              uint4* __restrict__ Aexp, long total) {
  long idx = (long)blockIdx.x * blockDim.x + threadIdx.x;
  if (idx >= total) return;
  int f = (int)(idx & (NFEAT - 1));
  float x = raw[idx];
  if (bias) x = silu_f(x + bias[f]);
  float s = silu_f(x);
  float bs[NBASES];
  spline_bases(x, bs);
  uint4 o;
  o.x = (unsigned)f2bf(s)     | ((unsigned)f2bf(bs[0]) << 16);
  o.y = (unsigned)f2bf(bs[1]) | ((unsigned)f2bf(bs[2]) << 16);
  o.z = (unsigned)f2bf(bs[3]) | ((unsigned)f2bf(bs[4]) << 16);
  o.w = (unsigned)f2bf(bs[5]) | ((unsigned)f2bf(bs[6]) << 16);
  Aexp[idx] = o;
}

// Wexp[o, i*8 + 0] = bw[o,i]; Wexp[o, i*8 + 1+t] = sw[o,i,t] * ss[o,i]
__global__ void build_wexp_kernel(const float* __restrict__ bw,
                                  const float* __restrict__ sw,
                                  const float* __restrict__ ss,
                                  uint4* __restrict__ Wexp, int total) {
  int idx = blockIdx.x * blockDim.x + threadIdx.x;   // over out*in
  if (idx >= total) return;
  float sc = ss[idx];
  const float* s = sw + (size_t)idx * NBASES;
  uint4 o;
  o.x = (unsigned)f2bf(bw[idx])    | ((unsigned)f2bf(s[0] * sc) << 16);
  o.y = (unsigned)f2bf(s[1] * sc)  | ((unsigned)f2bf(s[2] * sc) << 16);
  o.z = (unsigned)f2bf(s[3] * sc)  | ((unsigned)f2bf(s[4] * sc) << 16);
  o.w = (unsigned)f2bf(s[5] * sc)  | ((unsigned)f2bf(s[6] * sc) << 16);
  Wexp[idx] = o;
}

// out(M x 128) = Aexp(M x 1024) * Wexp^T(1024 x 128), bf16 WMMA, f32 accum.
// Block = 8 waves, one 64-column group; wave w -> M-tile (mb*8 + w).
// B (Wexp) k-chunks staged in LDS, shared by all 8 waves. When the async
// global->LDS path exists we double-buffer: stage chunk i+1 while WMMAs
// consume chunk i (tracked with ASYNCcnt), one barrier per chunk.
__global__ __launch_bounds__(256) void gemm_kernel(const __bf16* __restrict__ A,
                                                   const __bf16* __restrict__ W,
                                                   float* __restrict__ out,
                                                   int mtiles) {
#if HAVE_ASYNC_LDS
  __shared__ unsigned short Bsh[2 * BUFELEMS];   // 34 KB double buffer
#else
  __shared__ unsigned short Bsh[BUFELEMS];       // 17 KB single buffer
#endif

  const int group = blockIdx.x & 1;              // which half of the 128 output cols
  const int mb    = blockIdx.x >> 1;
  const int wave  = threadIdx.x >> 5;
  const int lane  = threadIdx.x & 31;
  const int mtile = mb * 8 + wave;
  const bool active = (mtile < mtiles);

  const int m  = lane & 15;
  const int hi = lane >> 4;
  const __bf16* arow = A + (size_t)(mtile * 16 + m) * KEXP;

  // cooperative staging assignment: 4 threads per column, 64 B each
  const int sc = threadIdx.x >> 2;               // column 0..63
  const int sq = threadIdx.x & 3;                // k quarter
  const uint4* gW = (const uint4*)(W + (size_t)(group * 64 + sc) * KEXP + sq * 32);
  unsigned short* sWbase = &Bsh[sc * BSTRIDE + sq * 32];

  v8f acc0 = {}, acc1 = {}, acc2 = {}, acc3 = {};

#if HAVE_ASYNC_LDS
  auto stage_async = [&](int kc, int b) {
    const uint4* g = gW + (kc >> 3);             // kc bf16 -> kc/8 uint4
    uint4* s = (uint4*)(sWbase + (size_t)b * BUFELEMS);
#pragma unroll
    for (int j = 0; j < 4; ++j)
      __builtin_amdgcn_global_load_async_to_lds_b128(
          (gptr_v4i)(unsigned long long)(g + j),
          (lptr_v4i)(unsigned long long)(s + j), 0, 0);
  };
  int buf = 0;
  stage_async(0, 0);
  __builtin_amdgcn_s_wait_asynccnt(0);
  __syncthreads();
#endif

  for (int kc = 0; kc < KEXP; kc += KC) {
#if HAVE_ASYNC_LDS
    if (kc + KC < KEXP) stage_async(kc + KC, buf ^ 1);
    const unsigned short* Bbuf = &Bsh[(size_t)buf * BUFELEMS];
#else
    {
      const uint4* g = gW + (kc >> 3);
      uint4* s = (uint4*)sWbase;
      s[0] = g[0]; s[1] = g[1]; s[2] = g[2]; s[3] = g[3];
    }
    __syncthreads();
    const unsigned short* Bbuf = &Bsh[0];
#endif
    if (active) {
#pragma unroll
      for (int kkl = 0; kkl < KC; kkl += 32) {
        const int kk = kc + kkl;
        // A 16x32 bf16 fragment: half hi -> K = kk+hi*16+[0..15] (contiguous, matches B)
        v8bf alo = *(const v8bf*)(arow + kk + hi * 16);
        v8bf ahi = *(const v8bf*)(arow + kk + hi * 16 + 8);
        v16bf a = __builtin_shufflevector(alo, ahi, 0,1,2,3,4,5,6,7,8,9,10,11,12,13,14,15);
        // B 32x16 fragment from LDS: lane (col m, half hi) -> 32 B contiguous
        v16bf b0 = *(const v16bf*)(&Bbuf[( 0 + m) * BSTRIDE + kkl + hi * 16]);
        v16bf b1 = *(const v16bf*)(&Bbuf[(16 + m) * BSTRIDE + kkl + hi * 16]);
        v16bf b2 = *(const v16bf*)(&Bbuf[(32 + m) * BSTRIDE + kkl + hi * 16]);
        v16bf b3 = *(const v16bf*)(&Bbuf[(48 + m) * BSTRIDE + kkl + hi * 16]);
        acc0 = __builtin_amdgcn_wmma_f32_16x16x32_bf16(false, a, false, b0, (short)0, acc0, false, false);
        acc1 = __builtin_amdgcn_wmma_f32_16x16x32_bf16(false, a, false, b1, (short)0, acc1, false, false);
        acc2 = __builtin_amdgcn_wmma_f32_16x16x32_bf16(false, a, false, b2, (short)0, acc2, false, false);
        acc3 = __builtin_amdgcn_wmma_f32_16x16x32_bf16(false, a, false, b3, (short)0, acc3, false, false);
      }
    }
#if HAVE_ASYNC_LDS
    __builtin_amdgcn_s_wait_asynccnt(0);
    __syncthreads();
    buf ^= 1;
#else
    __syncthreads();
#endif
  }

  if (!active) return;
  // C/D layout: VGPR r, lane -> row = hi*8 + r, col = m
  float* orow = out + (size_t)(mtile * 16 + hi * 8) * NFEAT + group * 64 + m;
#pragma unroll
  for (int r = 0; r < 8; ++r) {
    orow[(size_t)r * NFEAT +  0] = acc0[r];
    orow[(size_t)r * NFEAT + 16] = acc1[r];
    orow[(size_t)r * NFEAT + 32] = acc2[r];
    orow[(size_t)r * NFEAT + 48] = acc3[r];
  }
}

__global__ void deg_kernel(const int* __restrict__ dst, float* __restrict__ deg,
                           int E, int n) {
  int i = blockIdx.x * blockDim.x + threadIdx.x;
  if (i >= E + n) return;
  int d = (i < E) ? dst[i] : (i - E);     // self loops appended
  unsafeAtomicAdd(&deg[d], 1.0f);
}

__global__ void rsqrt_kernel(float* __restrict__ dinv, int n) {
  int i = blockIdx.x * blockDim.x + threadIdx.x;
  if (i >= n) return;
  float d = dinv[i];
  dinv[i] = (d > 0.0f) ? rsqrtf(fmaxf(d, 1e-12f)) : 0.0f;
}

// warp per (edge | self-loop): out[dst] += h[src] * dinv[src]*dinv[dst]
__global__ void aggregate_kernel(const float* __restrict__ h,
                                 const int* __restrict__ src,
                                 const int* __restrict__ dst,
                                 const float* __restrict__ dinv,
                                 float* __restrict__ out, int E, int n) {
  int wid  = (blockIdx.x * blockDim.x + threadIdx.x) >> 5;
  int lane = threadIdx.x & 31;
  if (wid >= E + n) return;
  int s, d;
  if (wid < E) { s = src[wid]; d = dst[wid]; } else { s = d = wid - E; }
  float w = dinv[s] * dinv[d];
  const float* hrow = h + (size_t)s * NFEAT;
  float* orow = out + (size_t)d * NFEAT;
#pragma unroll
  for (int j = 0; j < NFEAT; j += 32)
    unsafeAtomicAdd(&orow[lane + j], hrow[lane + j] * w);
}

__global__ void pool_kernel(const float* __restrict__ agg, const float* __restrict__ bias,
                            const int* __restrict__ batch,
                            float* __restrict__ pooled, float* __restrict__ counts,
                            long total) {
  long idx = (long)blockIdx.x * blockDim.x + threadIdx.x;
  if (idx >= total) return;
  int f = (int)(idx & (NFEAT - 1));
  int nidx = (int)(idx >> 7);
  int g = batch[nidx];
  float val = silu_f(agg[idx] + bias[f]);
  unsafeAtomicAdd(&pooled[g * NFEAT + f], val);
  if (f == 0) unsafeAtomicAdd(&counts[g], 1.0f);
}

__global__ void readout_kernel(const float* __restrict__ pooled, const float* __restrict__ counts,
                               const float* __restrict__ bwr, const float* __restrict__ swr,
                               const float* __restrict__ ssr, float* __restrict__ logits) {
  int idx = blockIdx.x * blockDim.x + threadIdx.x;    // NGRAPHS * NCLASSES
  if (idx >= NGRAPHS * NCLASSES) return;
  int g = idx / NCLASSES, c = idx % NCLASSES;
  float cnt = fmaxf(counts[g], 1.0f);
  float acc = 0.0f;
  for (int i = 0; i < NFEAT; ++i) {
    float p = pooled[g * NFEAT + i] / cnt;
    float s = silu_f(p);
    float bs[NBASES];
    spline_bases(p, bs);
    float sp = 0.0f;
    const float* swrow = swr + (size_t)(c * NFEAT + i) * NBASES;
#pragma unroll
    for (int t = 0; t < NBASES; ++t) sp += swrow[t] * bs[t];
    acc += bwr[c * NFEAT + i] * s + sp * ssr[c * NFEAT + i];
  }
  logits[idx] = acc;
}

__global__ void logsoftmax_kernel(float* __restrict__ out) {
  int g = blockIdx.x * blockDim.x + threadIdx.x;
  if (g >= NGRAPHS) return;
  float v[NCLASSES], m = -1e30f;
#pragma unroll
  for (int c = 0; c < NCLASSES; ++c) { v[c] = out[g * NCLASSES + c]; m = fmaxf(m, v[c]); }
  float s = 0.0f;
#pragma unroll
  for (int c = 0; c < NCLASSES; ++c) s += expf(v[c] - m);
  float l = logf(s);
#pragma unroll
  for (int c = 0; c < NCLASSES; ++c) out[g * NCLASSES + c] = v[c] - m - l;
}

extern "C" void kernel_launch(void* const* d_in, const int* in_sizes, int n_in,
                              void* d_out, int out_size, void* d_ws, size_t ws_size,
                              hipStream_t stream) {
  const float* x     = (const float*)d_in[0];
  const int*   ei    = (const int*)d_in[1];
  const int*   batch = (const int*)d_in[2];
  const float* bw[3] = {(const float*)d_in[3],  (const float*)d_in[7],  (const float*)d_in[11]};
  const float* sw[3] = {(const float*)d_in[4],  (const float*)d_in[8],  (const float*)d_in[12]};
  const float* ss[3] = {(const float*)d_in[5],  (const float*)d_in[9],  (const float*)d_in[13]};
  const float* bb[3] = {(const float*)d_in[6],  (const float*)d_in[10], (const float*)d_in[14]};
  const float* bwr = (const float*)d_in[15];
  const float* swr = (const float*)d_in[16];
  const float* ssr = (const float*)d_in[17];

  const int n = in_sizes[0] / NFEAT;   // 50000
  const int E = in_sizes[1] / 2;       // 800000
  const int* srcp = ei;
  const int* dstp = ei + E;

  char* ws = (char*)d_ws;
  size_t off = 0;
  auto carve = [&](size_t bytes) {
    void* p = (void*)(ws + off);
    off += (bytes + 255) & ~(size_t)255;
    return p;
  };
  unsigned short* Aexp = (unsigned short*)carve((size_t)n * KEXP * 2);       // ~102.4 MB
  unsigned short* Wexp = (unsigned short*)carve((size_t)NFEAT * KEXP * 2);   // 256 KB
  float* Flin   = (float*)carve((size_t)n * NFEAT * 4);                      // 25.6 MB
  float* Fagg   = (float*)carve((size_t)n * NFEAT * 4);                      // 25.6 MB
  float* dinv   = (float*)carve((size_t)n * 4);
  float* pooled = (float*)carve((size_t)NGRAPHS * NFEAT * 4);
  float* counts = (float*)carve((size_t)NGRAPHS * 4);
  (void)ws_size; (void)n_in; (void)out_size;

  const long total_nf = (long)n * NFEAT;
  const int TB = 256;

  // Degree / symmetric norm (depends only on graph) — once.
  (void)hipMemsetAsync(dinv, 0, (size_t)n * 4, stream);
  deg_kernel<<<(E + n + TB - 1) / TB, TB, 0, stream>>>(dstp, dinv, E, n);
  rsqrt_kernel<<<(n + TB - 1) / TB, TB, 0, stream>>>(dinv, n);

  const int mtiles  = n / 16;              // 3125
  const int mblocks = (mtiles + 7) / 8;    // 8 M-tiles per block
  for (int l = 0; l < 3; ++l) {
    const float* raw  = (l == 0) ? x : Fagg;
    const float* bias = (l == 0) ? nullptr : bb[l - 1];
    expand_kernel<<<(int)((total_nf + TB - 1) / TB), TB, 0, stream>>>(raw, bias, (uint4*)Aexp, total_nf);
    build_wexp_kernel<<<(NFEAT * NFEAT + TB - 1) / TB, TB, 0, stream>>>(bw[l], sw[l], ss[l], (uint4*)Wexp, NFEAT * NFEAT);
    gemm_kernel<<<mblocks * 2, TB, 0, stream>>>((const __bf16*)Aexp, (const __bf16*)Wexp, Flin, mtiles);
    (void)hipMemsetAsync(Fagg, 0, (size_t)n * NFEAT * 4, stream);
    {
      long warps  = (long)(E + n);
      long blocks = (warps + 7) / 8;
      aggregate_kernel<<<(int)blocks, TB, 0, stream>>>(Flin, srcp, dstp, dinv, Fagg, E, n);
    }
  }

  (void)hipMemsetAsync(pooled, 0, (size_t)NGRAPHS * NFEAT * 4, stream);
  (void)hipMemsetAsync(counts, 0, (size_t)NGRAPHS * 4, stream);
  pool_kernel<<<(int)((total_nf + TB - 1) / TB), TB, 0, stream>>>(Fagg, bb[2], batch, pooled, counts, total_nf);
  readout_kernel<<<(NGRAPHS * NCLASSES + TB - 1) / TB, TB, 0, stream>>>(pooled, counts, bwr, swr, ssr, (float*)d_out);
  logsoftmax_kernel<<<1, 64, 0, stream>>>((float*)d_out);
}
